// decoder_85169201480433
// MI455X (gfx1250) — matrix operand order repaired and checked by reading, not compiled, not run on hardware
//
#include <hip/hip_runtime.h>
#include <hip/hip_bf16.h>

// ---------------------------------------------------------------------------
// Dims (fixed by the reference)
#define TT 32
#define BB 16
#define DD 512
#define HH 512
#define EE 512
#define CC 512
#define FH 8
#define FW 32
#define HW 256            // FH*FW
#define G4 2048           // 4*H

typedef __attribute__((ext_vector_type(16))) __bf16 v16bf;
typedef __attribute__((ext_vector_type(8)))  __bf16 v8bf;
typedef __attribute__((ext_vector_type(8)))  float  v8f;

static __device__ __forceinline__ __bf16 f2bf(float f) {
    union { float f; unsigned u; } v; v.f = f;
    unsigned r = (v.u + 0x7FFFu + ((v.u >> 16) & 1u)) >> 16;
    unsigned short s = (unsigned short)r;
    __bf16 o; __builtin_memcpy(&o, &s, 2);
    return o;
}
static __device__ __forceinline__ float sigmoidf_(float x) {
    return 1.0f / (1.0f + __expf(-x));
}
#define WMMA_BF16(a, b, c) \
    __builtin_amdgcn_wmma_f32_16x16x32_bf16(false, (a), false, (b), (short)0, (c), false, false)

// A fragment (16x32 bf16, MxK), A row-major with leading dim lda.
static __device__ __forceinline__ v16bf load_a_frag(const __bf16* A, int lda,
                                                    int row_base, int k_base, int lane) {
    int m = row_base + (lane & 15);
    int k = k_base + ((lane >> 4) << 3);      // +8 halves for upper half-wave
    const __bf16* p = A + (size_t)m * lda + k;
    v8bf lo = *(const v8bf*)p;
    v8bf hi = *(const v8bf*)(p + 16);
    return __builtin_shufflevector(lo, hi, 0,1,2,3,4,5,6,7,8,9,10,11,12,13,14,15);
}
// B fragment (32x16 bf16, KxN) where weights are stored (N,K) row-major (== B col-major).
static __device__ __forceinline__ v16bf load_b_frag(const __bf16* W, int ldw,
                                                    int n_base, int k_base, int lane) {
    int n = n_base + (lane & 15);
    int k = k_base + ((lane >> 4) << 3);      // +8 halves for upper half-wave
    const __bf16* p = W + (size_t)n * ldw + k;
    v8bf lo = *(const v8bf*)p;
    v8bf hi = *(const v8bf*)(p + 16);
    return __builtin_shufflevector(lo, hi, 0,1,2,3,4,5,6,7,8,9,10,11,12,13,14,15);
}

// ---------------------------------------------------------------------------
// fp32 -> bf16 conversion
__global__ __launch_bounds__(256) void cvt_bf16_kernel(const float* __restrict__ src,
                                                       __bf16* __restrict__ dst, int n) {
    int i = blockIdx.x * 256 + threadIdx.x;
    if (i < n) dst[i] = f2bf(src[i]);
}

// conv_f (B,C,FH,FW) fp32 -> zero-padded NHWC bf16 (B, FH+2, FW+2, C)
__global__ __launch_bounds__(256) void pad_nhwc_kernel(const float* __restrict__ src,
                                                       __bf16* __restrict__ dst) {
    int idx = blockIdx.x * 256 + threadIdx.x;
    const int total = BB * (FH + 2) * (FW + 2) * CC;
    if (idx >= total) return;
    int c  = idx & (CC - 1);
    int r  = idx >> 9;
    int xx = r % (FW + 2); r /= (FW + 2);
    int yy = r % (FH + 2); int b = r / (FH + 2);
    int y = yy - 1, x = xx - 1;
    float v = 0.f;
    if (y >= 0 && y < FH && x >= 0 && x < FW)
        v = src[(((size_t)b * CC + c) * FH + y) * FW + x];
    dst[idx] = f2bf(v);
}

// W_em (E,C,3,3) OIHW fp32 -> (tap, E, C) bf16 (tap = dy*3+dx)
__global__ __launch_bounds__(256) void cvt_wem_kernel(const float* __restrict__ src,
                                                      __bf16* __restrict__ dst) {
    int idx = blockIdx.x * 256 + threadIdx.x;
    const int total = 9 * EE * CC;
    if (idx >= total) return;
    int c = idx & (CC - 1);
    int r = idx >> 9;
    int e = r & (EE - 1);
    int tap = r >> 9;
    int dy = tap / 3, dx = tap % 3;
    dst[idx] = f2bf(src[(((size_t)e * CC + c) * 3 + dy) * 3 + dx]);
}

// ---------------------------------------------------------------------------
// Generic WMMA GEMM with 2x2 register blocking (32x32 macro-tile per wave):
//   C[M,N] (f32) = A[M,K](bf16, row-major) * B[N,K](bf16)^T + bias0 + bias1
// M, N must be multiples of 32.
__global__ __launch_bounds__(256) void wmma_gemm_nt(const __bf16* __restrict__ A, int lda,
                                                    const __bf16* __restrict__ Bw, int ldb,
                                                    const float* __restrict__ bias0,
                                                    const float* __restrict__ bias1,
                                                    float* __restrict__ C, int ldc,
                                                    int M, int N, int K) {
    int wid  = (blockIdx.x * 256 + threadIdx.x) >> 5;
    int lane = threadIdx.x & 31;
    int nblk = N >> 5;
    int blocks = (M >> 5) * nblk;
    if (wid >= blocks) return;                // uniform per wave -> EXEC stays all-1s
    int mb = wid / nblk, nb = wid % nblk;
    int m0 = mb << 5, n0 = nb << 5;
    v8f acc00 = {}, acc01 = {}, acc10 = {}, acc11 = {};
    for (int k = 0; k < K; k += 32) {
        v16bf a0 = load_a_frag(A, lda, m0,      k, lane);
        v16bf a1 = load_a_frag(A, lda, m0 + 16, k, lane);
        v16bf b0 = load_b_frag(Bw, ldb, n0,      k, lane);
        v16bf b1 = load_b_frag(Bw, ldb, n0 + 16, k, lane);
        acc00 = WMMA_BF16(a0, b0, acc00);
        acc01 = WMMA_BF16(a0, b1, acc01);
        acc10 = WMMA_BF16(a1, b0, acc10);
        acc11 = WMMA_BF16(a1, b1, acc11);
    }
    int nc0 = n0 + (lane & 15), nc1 = nc0 + 16;
    float bi0 = (bias0 ? bias0[nc0] : 0.f) + (bias1 ? bias1[nc0] : 0.f);
    float bi1 = (bias0 ? bias0[nc1] : 0.f) + (bias1 ? bias1[nc1] : 0.f);
    int mr0 = m0 + ((lane >> 4) << 3), mr1 = mr0 + 16;
    #pragma unroll
    for (int r = 0; r < 8; ++r) {
        C[(size_t)(mr0 + r) * ldc + nc0] = acc00[r] + bi0;
        C[(size_t)(mr0 + r) * ldc + nc1] = acc01[r] + bi1;
        C[(size_t)(mr1 + r) * ldc + nc0] = acc10[r] + bi0;
        C[(size_t)(mr1 + r) * ldc + nc1] = acc11[r] + bi1;
    }
}

// ---------------------------------------------------------------------------
// Bidirectional LSTM recurrence. grid = 2 blocks (dir), block = 256 (8 waves).
// gx already contains x*W_ih^T + b_ih + b_hh.  h kept in LDS (bf16), c in regs.
// Wave w owns hidden units [w*64, w*64+64) as 4 groups of 16 across all 4 gates.
// A-fragment (h) is loaded once per K-chunk and reused across the 4 gates.
__global__ __launch_bounds__(256) void lstm_kernel(const float* __restrict__ gx_f,
                                                   const float* __restrict__ gx_b,
                                                   const __bf16* __restrict__ Whh_f,
                                                   const __bf16* __restrict__ Whh_b,
                                                   __bf16* __restrict__ h_cat) {
    const int dir = blockIdx.x;
    const float*  gx  = dir ? gx_b  : gx_f;
    const __bf16* Whh = dir ? Whh_b : Whh_f;
    const int tid  = threadIdx.x;
    const int lane = tid & 31;
    const int wave = tid >> 5;

    __shared__ __bf16 h_lds[16][520];          // padded stride to dodge bank conflicts

    for (int i = tid; i < 16 * 520; i += 256) (&h_lds[0][0])[i] = f2bf(0.f);
    __syncthreads();

    const int lr   = lane & 15;                // column within tile / A row
    const int hb   = (lane >> 4) << 3;         // +8 row offset for upper half-wave
    const int hb16 = (lane >> 4) << 3;         // +8 halves for B frag
    const int jb0  = wave << 6;                // hidden-unit base of this wave

    v8f cstate[4]; v8f hstage[4];
    #pragma unroll
    for (int g = 0; g < 4; ++g) { cstate[g] = (v8f){}; hstage[g] = (v8f){}; }

    for (int s = 0; s < TT; ++s) {
        int t = dir ? (TT - 1 - s) : s;
        const float* gxt = gx + (size_t)t * BB * G4;

        for (int grp = 0; grp < 4; ++grp) {
            int jb = jb0 + (grp << 4);
            v8f acc[4];
            const __bf16* wrow[4];
            #pragma unroll
            for (int gate = 0; gate < 4; ++gate) {
                int n = gate * HH + jb + lr;    // row in W_hh / column in gates
                wrow[gate] = Whh + (size_t)n * HH + hb16;
                v8f a;
                #pragma unroll
                for (int r = 0; r < 8; ++r)
                    a[r] = gxt[(size_t)(r + hb) * G4 + n];
                acc[gate] = a;
            }
            for (int k = 0; k < HH; k += 32) {
                const __bf16* pa = &h_lds[lr][k + hb];
                v8bf lo = *(const v8bf*)pa;
                v8bf hi = *(const v8bf*)(pa + 16);
                v16bf af = __builtin_shufflevector(lo, hi,
                           0,1,2,3,4,5,6,7,8,9,10,11,12,13,14,15);
                #pragma unroll
                for (int gate = 0; gate < 4; ++gate) {
                    v8bf blo = *(const v8bf*)(wrow[gate] + k);
                    v8bf bhi = *(const v8bf*)(wrow[gate] + k + 16);
                    v16bf bf = __builtin_shufflevector(blo, bhi, 0,1,2,3,4,5,6,7,8,9,10,11,12,13,14,15);
                    acc[gate] = WMMA_BF16(af, bf, acc[gate]);
                }
            }
            v8f cg = cstate[grp]; v8f hv;
            #pragma unroll
            for (int r = 0; r < 8; ++r) {
                float iv = sigmoidf_(acc[0][r]);
                float fv = sigmoidf_(acc[1][r]);
                float gv = tanhf(acc[2][r]);
                float ov = sigmoidf_(acc[3][r]);
                float cn = fv * cg[r] + iv * gv;
                cg[r] = cn;
                hv[r] = ov * tanhf(cn);
            }
            cstate[grp] = cg; hstage[grp] = hv;
            #pragma unroll
            for (int r = 0; r < 8; ++r) {
                int b = r + hb;
                h_cat[(size_t)(t * BB + b) * (2 * HH) + dir * HH + jb + lr] = f2bf(hv[r]);
            }
        }
        __syncthreads();                       // all waves done reading h_lds
        for (int grp = 0; grp < 4; ++grp) {
            int jb = jb0 + (grp << 4);
            #pragma unroll
            for (int r = 0; r < 8; ++r)
                h_lds[r + hb][jb + lr] = f2bf(hstage[grp][r]);
        }
        __syncthreads();
    }
}

// ---------------------------------------------------------------------------
// hidden_de (T*B,H) f32 -> out[b][t][0:H] and bf16 copy for h_em GEMM
__global__ __launch_bounds__(256) void scatter_hde_kernel(const float* __restrict__ hde,
                                                          float* __restrict__ out,
                                                          __bf16* __restrict__ hde_bf) {
    int idx = blockIdx.x * 256 + threadIdx.x;
    if (idx >= TT * BB * HH) return;
    int j = idx & (HH - 1);
    int r = idx >> 9;
    int b = r & (BB - 1);
    int t = r >> 4;
    float v = hde[idx];
    out[((size_t)b * TT + t) * (HH + EE) + j] = v;
    hde_bf[idx] = f2bf(v);
}

// ---------------------------------------------------------------------------
// 3x3 conv as 9-tap implicit GEMM over padded NHWC input, 2x2 register blocking.
// An M-block of 32 is one full (b,y) image row (x = 0..31).
// Y (B*HW, E) f32 = sum_tap Xpad(b,y+dy,x+dx,:) * Wt(tap,:,:)^T + bias
__global__ __launch_bounds__(256) void conv_wmma_kernel(const __bf16* __restrict__ Xp,
                                                        const __bf16* __restrict__ Wt,
                                                        const float* __restrict__ bias,
                                                        float* __restrict__ Y) {
    int wid  = (blockIdx.x * 256 + threadIdx.x) >> 5;
    int lane = threadIdx.x & 31;
    int nb = wid & 15;                 // 16 blocks over E   (32 wide)
    int mb = wid >> 4;                 // 128 blocks over B*HW (32 wide)
    int m0 = mb << 5;
    int m  = m0 + (lane & 15);         // first tile row; second is m+16
    int b = m >> 8;
    int y = (m >> 5) & 7;
    int x = m & 31;                    // 0..15 (m0 is 32-aligned)
    int hb8  = (lane >> 4) << 3;
    int hb16 = (lane >> 4) << 3;
    int n0 = nb << 5;
    int n  = n0 + (lane & 15);
    v8f acc00 = {}, acc01 = {}, acc10 = {}, acc11 = {};
    for (int tap = 0; tap < 9; ++tap) {
        int dy = tap / 3, dx = tap % 3;
        const __bf16* pa0 = Xp + ((size_t)((b * (FH + 2) + y + dy) * (FW + 2) + (x + dx))) * CC + hb8;
        const __bf16* pa1 = pa0 + (size_t)16 * CC;           // x+16 in same padded row
        const __bf16* pb0 = Wt + ((size_t)(tap * EE + n)) * CC + hb16;
        const __bf16* pb1 = pb0 + (size_t)16 * CC;           // n+16
        for (int k = 0; k < CC; k += 32) {
            v8bf lo0 = *(const v8bf*)(pa0 + k);
            v8bf hi0 = *(const v8bf*)(pa0 + k + 16);
            v16bf a0 = __builtin_shufflevector(lo0, hi0,
                       0,1,2,3,4,5,6,7,8,9,10,11,12,13,14,15);
            v8bf lo1 = *(const v8bf*)(pa1 + k);
            v8bf hi1 = *(const v8bf*)(pa1 + k + 16);
            v16bf a1 = __builtin_shufflevector(lo1, hi1,
                       0,1,2,3,4,5,6,7,8,9,10,11,12,13,14,15);
            v16bf b0 = __builtin_shufflevector(*(const v8bf*)(pb0 + k), *(const v8bf*)(pb0 + k + 16),
                       0,1,2,3,4,5,6,7,8,9,10,11,12,13,14,15);
            v16bf b1 = __builtin_shufflevector(*(const v8bf*)(pb1 + k), *(const v8bf*)(pb1 + k + 16),
                       0,1,2,3,4,5,6,7,8,9,10,11,12,13,14,15);
            acc00 = WMMA_BF16(a0, b0, acc00);
            acc01 = WMMA_BF16(a0, b1, acc01);
            acc10 = WMMA_BF16(a1, b0, acc10);
            acc11 = WMMA_BF16(a1, b1, acc11);
        }
    }
    int nc0 = n0 + (lane & 15), nc1 = nc0 + 16;
    float bi0 = bias[nc0], bi1 = bias[nc1];
    int mr0 = m0 + hb8, mr1 = mr0 + 16;
    #pragma unroll
    for (int r = 0; r < 8; ++r) {
        Y[(size_t)(mr0 + r) * EE + nc0] = acc00[r] + bi0;
        Y[(size_t)(mr0 + r) * EE + nc1] = acc01[r] + bi1;
        Y[(size_t)(mr1 + r) * EE + nc0] = acc10[r] + bi0;
        Y[(size_t)(mr1 + r) * EE + nc1] = acc11[r] + bi1;
    }
}

// ---------------------------------------------------------------------------
// Attention: one block per (t,b). scores -> softmax -> alpha * x_em, writes out[..,H:H+E].
__global__ __launch_bounds__(256) void attention_kernel(const float* __restrict__ xem,  // (B,HW,E)
                                                        const float* __restrict__ hem,  // (T*B,E)
                                                        const float* __restrict__ w_att,
                                                        const float* __restrict__ b_att,
                                                        float* __restrict__ out) {
    int t = blockIdx.x >> 4;
    int b = blockIdx.x & 15;
    int tid = threadIdx.x;
    __shared__ float hvec[EE];
    __shared__ float wv[EE];
    __shared__ float sc[HW];
    __shared__ float red[HW];
    const float* hr = hem + ((size_t)t * BB + b) * EE;
    hvec[tid] = hr[tid]; hvec[tid + 256] = hr[tid + 256];
    wv[tid] = w_att[tid]; wv[tid + 256] = w_att[tid + 256];
    __syncthreads();

    const float* xr = xem + ((size_t)b * HW + tid) * EE;
    float s = 0.f;
    for (int e = 0; e < EE; ++e) s += tanhf(xr[e] + hvec[e]) * wv[e];
    s += b_att[0];
    sc[tid] = s; red[tid] = s;
    __syncthreads();
    for (int off = 128; off > 0; off >>= 1) {
        if (tid < off) red[tid] = fmaxf(red[tid], red[tid + off]);
        __syncthreads();
    }
    float mx = red[0];
    __syncthreads();
    float ex = __expf(s - mx);
    sc[tid] = ex; red[tid] = ex;
    __syncthreads();
    for (int off = 128; off > 0; off >>= 1) {
        if (tid < off) red[tid] += red[tid + off];
        __syncthreads();
    }
    float inv = 1.f / red[0];
    __syncthreads();

    float a0 = 0.f, a1 = 0.f;
    for (int hw = 0; hw < HW; ++hw) {
        float al = sc[hw] * inv;
        const float* row = xem + ((size_t)b * HW + hw) * EE;
        a0 += al * row[tid];
        a1 += al * row[tid + 256];
    }
    float* o = out + ((size_t)b * TT + t) * (HH + EE) + HH;
    o[tid] = a0; o[tid + 256] = a1;
}

// ---------------------------------------------------------------------------
extern "C" void kernel_launch(void* const* d_in, const int* in_sizes, int n_in,
                              void* d_out, int out_size, void* d_ws, size_t ws_size,
                              hipStream_t stream) {
    (void)in_sizes; (void)n_in; (void)out_size; (void)ws_size;

    const float* hidden_en = (const float*)d_in[0];
    const float* conv_f    = (const float*)d_in[1];
    const float* W_ih_f    = (const float*)d_in[2];
    const float* W_hh_f    = (const float*)d_in[3];
    const float* b_ih_f    = (const float*)d_in[4];
    const float* b_hh_f    = (const float*)d_in[5];
    const float* W_ih_b    = (const float*)d_in[6];
    const float* W_hh_b    = (const float*)d_in[7];
    const float* b_ih_b    = (const float*)d_in[8];
    const float* b_hh_b    = (const float*)d_in[9];
    const float* W_dec     = (const float*)d_in[10];
    const float* b_dec     = (const float*)d_in[11];
    const float* W_em      = (const float*)d_in[12];
    const float* b_em      = (const float*)d_in[13];
    const float* W_hem     = (const float*)d_in[14];
    const float* b_hem     = (const float*)d_in[15];
    const float* w_att     = (const float*)d_in[16];
    const float* b_att     = (const float*)d_in[17];
    float* out = (float*)d_out;

    // workspace carve-up
    char* w = (char*)d_ws;
    auto carve = [&](size_t bytes) { void* p = (void*)w; w += (bytes + 255) & ~(size_t)255; return p; };
    __bf16* x_bf     = (__bf16*)carve((size_t)TT * BB * DD * 2);
    __bf16* Wihf_bf  = (__bf16*)carve((size_t)G4 * DD * 2);
    __bf16* Whhf_bf  = (__bf16*)carve((size_t)G4 * HH * 2);
    __bf16* Wihb_bf  = (__bf16*)carve((size_t)G4 * DD * 2);
    __bf16* Whhb_bf  = (__bf16*)carve((size_t)G4 * HH * 2);
    float*  gx_f     = (float*) carve((size_t)TT * BB * G4 * 4);
    float*  gx_b     = (float*) carve((size_t)TT * BB * G4 * 4);
    __bf16* hcat_bf  = (__bf16*)carve((size_t)TT * BB * 2 * HH * 2);
    __bf16* Wdec_bf  = (__bf16*)carve((size_t)HH * 2 * HH * 2);
    float*  hde_f32  = (float*) carve((size_t)TT * BB * HH * 4);
    __bf16* hde_bf   = (__bf16*)carve((size_t)TT * BB * HH * 2);
    __bf16* Whem_bf  = (__bf16*)carve((size_t)EE * HH * 2);
    float*  hem_f32  = (float*) carve((size_t)TT * BB * EE * 4);
    __bf16* xpad_bf  = (__bf16*)carve((size_t)BB * (FH + 2) * (FW + 2) * CC * 2);
    __bf16* wemt_bf  = (__bf16*)carve((size_t)9 * EE * CC * 2);
    float*  xem_f32  = (float*) carve((size_t)BB * HW * EE * 4);

    auto cvt = [&](const float* s, __bf16* d, int n) {
        cvt_bf16_kernel<<<(n + 255) / 256, 256, 0, stream>>>(s, d, n);
    };
    // conversions
    cvt(hidden_en, x_bf,    TT * BB * DD);
    cvt(W_ih_f,    Wihf_bf, G4 * DD);
    cvt(W_hh_f,    Whhf_bf, G4 * HH);
    cvt(W_ih_b,    Wihb_bf, G4 * DD);
    cvt(W_hh_b,    Whhb_bf, G4 * HH);
    cvt(W_dec,     Wdec_bf, HH * 2 * HH);
    cvt(W_hem,     Whem_bf, EE * HH);
    {
        int n = BB * (FH + 2) * (FW + 2) * CC;
        pad_nhwc_kernel<<<(n + 255) / 256, 256, 0, stream>>>(conv_f, xpad_bf);
        int m = 9 * EE * CC;
        cvt_wem_kernel<<<(m + 255) / 256, 256, 0, stream>>>(W_em, wemt_bf);
    }

    // gx = x * W_ih^T + b_ih + b_hh   (both directions)
    {
        int waves = (TT * BB / 32) * (G4 / 32);                  // 1024 waves
        wmma_gemm_nt<<<waves / 8, 256, 0, stream>>>(x_bf, DD, Wihf_bf, DD,
                                                    b_ih_f, b_hh_f, gx_f, G4,
                                                    TT * BB, G4, DD);
        wmma_gemm_nt<<<waves / 8, 256, 0, stream>>>(x_bf, DD, Wihb_bf, DD,
                                                    b_ih_b, b_hh_b, gx_b, G4,
                                                    TT * BB, G4, DD);
    }

    // recurrence (fwd + bwd in one kernel, 2 workgroups)
    lstm_kernel<<<2, 256, 0, stream>>>(gx_f, gx_b, Whhf_bf, Whhb_bf, hcat_bf);

    // hidden_de = hcat * W_dec^T + b_dec
    {
        int waves = (TT * BB / 32) * (HH / 32);                  // 256 waves
        wmma_gemm_nt<<<waves / 8, 256, 0, stream>>>(hcat_bf, 2 * HH, Wdec_bf, 2 * HH,
                                                    b_dec, nullptr, hde_f32, HH,
                                                    TT * BB, HH, 2 * HH);
    }
    {
        int n = TT * BB * HH;
        scatter_hde_kernel<<<(n + 255) / 256, 256, 0, stream>>>(hde_f32, out, hde_bf);
    }

    // h_em = hidden_de * W_hem^T + b_hem
    {
        int waves = (TT * BB / 32) * (EE / 32);
        wmma_gemm_nt<<<waves / 8, 256, 0, stream>>>(hde_bf, HH, Whem_bf, HH,
                                                    b_hem, nullptr, hem_f32, EE,
                                                    TT * BB, EE, HH);
    }

    // conv_em as implicit GEMM  -> xem (B,HW,E) f32 (+ b_em)
    {
        int waves = (BB * HW / 32) * (EE / 32);                  // 2048 waves
        conv_wmma_kernel<<<waves / 8, 256, 0, stream>>>(xpad_bf, wemt_bf, b_em, xem_f32);
    }

    // attention + write att half of the output
    attention_kernel<<<TT * BB, 256, 0, stream>>>(xem_f32, hem_f32, w_att, b_att, out);
}
